// DeepseekV2MoE_CPP_44848048505224
// MI455X (gfx1250) — compile-verified, matrix operand order checked by reading, not run on hardware
//
#include <hip/hip_runtime.h>
#include <hip/hip_bf16.h>
#include <math.h>

typedef __bf16 bf16_t;
typedef __attribute__((ext_vector_type(16))) __bf16 v16bf;
typedef __attribute__((ext_vector_type(8)))  float  v8f;

#define T_TOK  512
#define H_DIM  2048
#define NEXP   32
#define TOPK   6
#define F_DIM  1408
#define FS_DIM 2816
#define CAP    160

#define BM 32
#define BN 128
#define BK 32
#define LDK  40   // bf16 A-tile LDS K-stride (padded)
#define LDKF 36   // fp32 B-tile LDS K-stride (padded, 144B rows)

// ---------------- CDNA5 async copy primitives ----------------

// GLOBAL_LOAD_ASYNC_TO_LDS_B128 (GV mode): per-lane 16B memory->LDS DMA,
// tracked by ASYNCcnt. vdst = LDS byte address, vaddr = 64-bit global address.
__device__ __forceinline__ void async_load_b128(unsigned lds_off, const void* gptr) {
  asm volatile("global_load_async_to_lds_b128 %0, %1, off"
               :: "v"(lds_off), "v"((unsigned long long)(uintptr_t)gptr)
               : "memory");
}

template <int N>
__device__ __forceinline__ void wait_async() {
  asm volatile("s_wait_asynccnt %0" :: "n"(N) : "memory");
}

// wave-cooperative: wave w async-copies its own 16 rows (fp32, 32 cols) of a
// B tile; 4 instructions x 32 lanes x 16B = 2KB per wave.
__device__ __forceinline__ void issue_B_async(float* BsBuf, const float* gbase,
                                              size_t ldg, int k0, int w, int lane) {
  const int rloc = lane >> 3;          // 0..3
  const int c4   = (lane & 7) << 2;    // 0,4,...,28
#pragma unroll
  for (int j = 0; j < 4; ++j) {
    const int row = w * 16 + j * 4 + rloc;
    const unsigned lds = (unsigned)(uintptr_t)(BsBuf + row * LDKF + c4);
    async_load_b128(lds, gbase + (size_t)row * ldg + k0 + c4);
  }
}

// ---------------- fragment builders ----------------

__device__ __forceinline__ void cvt_store4(bf16_t* dst, float4 v) {
  dst[0] = (bf16_t)v.x; dst[1] = (bf16_t)v.y;
  dst[2] = (bf16_t)v.z; dst[3] = (bf16_t)v.w;
}

// bf16 LDS tile [rows][LDK]: lane L holds row row0+L%16, K slots per the CDNA5
// 16-bit layout (lanes 0-15: K {0..7,16..23}; lanes 16-31: K {8..15,24..31}).
__device__ __forceinline__ v16bf frag_from_lds(const bf16_t* base, int row0, int lane) {
  const int r  = row0 + (lane & 15);
  const int kb = (lane >> 4) << 3;
  union { v16bf v; uint4 q[2]; } f;
  f.q[0] = *(const uint4*)(base + r * LDK + kb);
  f.q[1] = *(const uint4*)(base + r * LDK + 16 + kb);
  return f.v;
}

// fp32 LDS tile [rows][LDKF] -> bf16 fragment (convert at read; each B element
// is consumed by exactly one wave).
__device__ __forceinline__ v16bf frag_from_lds_f32(const float* base, int row0, int lane) {
  const int r  = row0 + (lane & 15);
  const int kb = (lane >> 4) << 3;
  const float* p0 = base + r * LDKF + kb;
  const float* p1 = base + r * LDKF + 16 + kb;
  union { v16bf v; bf16_t e[16]; } f;
#pragma unroll
  for (int j = 0; j < 8; ++j) f.e[j] = (bf16_t)p0[j];
#pragma unroll
  for (int j = 0; j < 8; ++j) f.e[8 + j] = (bf16_t)p1[j];
  return f.v;
}

__device__ __forceinline__ v8f wmma_bf16(v16bf a, v16bf b, v8f c) {
  return __builtin_amdgcn_wmma_f32_16x16x32_bf16(false, a, false, b, (short)0, c, false, false);
}

__device__ __forceinline__ float silu_mul(float g, float u) {
  return (g / (1.0f + __expf(-g))) * u;
}

// ---------------- kernel 0: zero the routing counters ----------------

__global__ void moe_zero_counts(int* counts) {
  if (threadIdx.x < NEXP) counts[threadIdx.x] = 0;
}

// ---------------- kernel 1: gate (softmax + top-6 + dispatch) ----------------

__global__ __launch_bounds__(32)
void moe_gate(const float* __restrict__ x, const float* __restrict__ wg,
              int* __restrict__ counts, int* __restrict__ expert_tok,
              int* __restrict__ slot_map, float* __restrict__ topkw) {
  const int t = blockIdx.x;
  const int e = threadIdx.x;
  const float4* xr = (const float4*)(x  + (size_t)t * H_DIM);
  const float4* wr = (const float4*)(wg + (size_t)e * H_DIM);
  float acc = 0.0f;
  for (int i = 0; i < H_DIM / 4; ++i) {
    float4 a = xr[i], b = wr[i];
    acc += a.x * b.x + a.y * b.y + a.z * b.z + a.w * b.w;
  }
  float m = acc;
  for (int off = 16; off; off >>= 1) m = fmaxf(m, __shfl_xor(m, off));
  float p = __expf(acc - m);
  float s = p;
  for (int off = 16; off; off >>= 1) s += __shfl_xor(s, off);
  float score = p / s;
  for (int k = 0; k < TOPK; ++k) {
    float v = score; int bi = e;
    for (int off = 16; off; off >>= 1) {
      float ov = __shfl_xor(v, off);
      int   oi = __shfl_xor(bi, off);
      if (ov > v || (ov == v && oi < bi)) { v = ov; bi = oi; }
    }
    if (e == bi) score = -1.0f;
    if (e == 0) {
      int pos  = atomicAdd(&counts[bi], 1);
      int slot = (pos < CAP) ? (bi * CAP + pos) : (NEXP * CAP);
      if (pos < CAP) expert_tok[bi * CAP + pos] = t;
      slot_map[t * TOPK + k] = slot;
      topkw[t * TOPK + k]    = v;
    }
  }
}

// ---------------- kernel 2: routed gate/up GEMM + SiLU*up -> h (bf16) -------
// grid (F/BN, CAP/BM, E); 8 waves; double-buffered, async B staging

__global__ __launch_bounds__(256)
void moe_gateup_routed(const float* __restrict__ x,
                       const float* __restrict__ gate_w,
                       const float* __restrict__ up_w,
                       const int* __restrict__ counts,
                       const int* __restrict__ expert_tok,
                       bf16_t* __restrict__ hbuf) {
  __shared__ bf16_t As[2][BM * LDK];
  __shared__ float  Bg[2][BN * LDKF];
  __shared__ float  Bu[2][BN * LDKF];
  const int e    = blockIdx.z;
  const int m0   = blockIdx.y * BM;
  const int n0   = blockIdx.x * BN;
  const int tid  = threadIdx.x;
  const int lane = tid & 31;
  const int w    = tid >> 5;
  const int cnt  = min(counts[e], CAP);
  const int arow = tid >> 3;
  const int acol = (tid & 7) << 2;

  const float amask = (m0 + arow < cnt) ? 1.0f : 0.0f;
  const int   tok   = (m0 + arow < cnt) ? expert_tok[e * CAP + m0 + arow] : 0;
  const float* xrow = x + (size_t)tok * H_DIM;
  const float* gbase = gate_w + ((size_t)e * F_DIM + n0) * H_DIM;
  const float* ubase = up_w   + ((size_t)e * F_DIM + n0) * H_DIM;

  // prologue: stage slab 0
  issue_B_async(Bg[0], gbase, H_DIM, 0, w, lane);
  issue_B_async(Bu[0], ubase, H_DIM, 0, w, lane);
  {
    float4 a0 = *(const float4*)(xrow + acol);
    a0.x *= amask; a0.y *= amask; a0.z *= amask; a0.w *= amask;
    cvt_store4(&As[0][arow * LDK + acol], a0);
  }

  v8f accg0 = {}, accg1 = {}, accu0 = {}, accu1 = {};
  const int KITER = H_DIM / BK;
  for (int it = 0; it < KITER; ++it) {
    const int buf = it & 1;
    const int k0  = it * BK;
    const bool have_next = (it + 1 < KITER);
    float4 areg = make_float4(0.f, 0.f, 0.f, 0.f);
    if (have_next) {
      issue_B_async(Bg[buf ^ 1], gbase, H_DIM, k0 + BK, w, lane);
      issue_B_async(Bu[buf ^ 1], ubase, H_DIM, k0 + BK, w, lane);
      areg = *(const float4*)(xrow + k0 + BK + acol);
      areg.x *= amask; areg.y *= amask; areg.z *= amask; areg.w *= amask;
    }
    __syncthreads();                              // As[buf] visible
    if (have_next) wait_async<8>(); else wait_async<0>();  // Bs[buf] landed

    v16bf a0 = frag_from_lds(As[buf], 0, lane);
    v16bf a1 = frag_from_lds(As[buf], 16, lane);
    v16bf bg = frag_from_lds_f32(Bg[buf], w * 16, lane);
    v16bf bu = frag_from_lds_f32(Bu[buf], w * 16, lane);
    accg0 = wmma_bf16(a0, bg, accg0);
    accg1 = wmma_bf16(a1, bg, accg1);
    accu0 = wmma_bf16(a0, bu, accu0);
    accu1 = wmma_bf16(a1, bu, accu1);

    if (have_next) cvt_store4(&As[buf ^ 1][arow * LDK + acol], areg);
  }

  const int nCol = n0 + w * 16 + (lane & 15);
  const int rb   = (lane >> 4) * 8;
#pragma unroll
  for (int r = 0; r < 8; ++r) {
    hbuf[((size_t)e * CAP + m0 + rb + r)      * F_DIM + nCol] = (bf16_t)silu_mul(accg0[r], accu0[r]);
    hbuf[((size_t)e * CAP + m0 + 16 + rb + r) * F_DIM + nCol] = (bf16_t)silu_mul(accg1[r], accu1[r]);
  }
}

// ---------------- kernel 3: routed down GEMM -> ye (fp32) ----------------

__global__ __launch_bounds__(256)
void moe_down_routed(const bf16_t* __restrict__ hbuf,
                     const float* __restrict__ down_w,
                     float* __restrict__ ye) {
  __shared__ bf16_t As[2][BM * LDK];
  __shared__ float  Bs[2][BN * LDKF];
  const int e    = blockIdx.z;
  const int m0   = blockIdx.y * BM;
  const int n0   = blockIdx.x * BN;
  const int tid  = threadIdx.x;
  const int lane = tid & 31;
  const int w    = tid >> 5;
  const int arow = tid >> 3;
  const int acol = (tid & 7) << 2;

  const bf16_t* hrow  = hbuf   + ((size_t)e * CAP + m0 + arow) * F_DIM;
  const float*  bbase = down_w + ((size_t)e * H_DIM + n0) * F_DIM;

  issue_B_async(Bs[0], bbase, F_DIM, 0, w, lane);
  *(uint2*)&As[0][arow * LDK + acol] = *(const uint2*)(hrow + acol);

  v8f acc0 = {}, acc1 = {};
  const int KITER = F_DIM / BK;
  for (int it = 0; it < KITER; ++it) {
    const int buf = it & 1;
    const int k0  = it * BK;
    const bool have_next = (it + 1 < KITER);
    uint2 areg = make_uint2(0u, 0u);
    if (have_next) {
      issue_B_async(Bs[buf ^ 1], bbase, F_DIM, k0 + BK, w, lane);
      areg = *(const uint2*)(hrow + k0 + BK + acol);
    }
    __syncthreads();
    if (have_next) wait_async<4>(); else wait_async<0>();

    v16bf a0 = frag_from_lds(As[buf], 0, lane);
    v16bf a1 = frag_from_lds(As[buf], 16, lane);
    v16bf b  = frag_from_lds_f32(Bs[buf], w * 16, lane);
    acc0 = wmma_bf16(a0, b, acc0);
    acc1 = wmma_bf16(a1, b, acc1);

    if (have_next) *(uint2*)&As[buf ^ 1][arow * LDK + acol] = areg;
  }

  const int n  = n0 + w * 16 + (lane & 15);
  const int rb = (lane >> 4) * 8;
#pragma unroll
  for (int r = 0; r < 8; ++r) {
    ye[((size_t)e * CAP + m0 + rb + r)      * H_DIM + n] = acc0[r];
    ye[((size_t)e * CAP + m0 + 16 + rb + r) * H_DIM + n] = acc1[r];
  }
}

// ---------------- kernel 4: combine (deterministic gather) ----------------

__global__ __launch_bounds__(256)
void moe_combine(const float* __restrict__ ye, const int* __restrict__ slot_map,
                 const float* __restrict__ topkw, float* __restrict__ out) {
  const int t = blockIdx.x;
  int   slots[TOPK];
  float ww[TOPK];
#pragma unroll
  for (int k = 0; k < TOPK; ++k) {
    slots[k] = slot_map[t * TOPK + k];
    ww[k]    = topkw[t * TOPK + k];
  }
  for (int h = threadIdx.x; h < H_DIM; h += 256) {
    float acc = 0.0f;
#pragma unroll
    for (int k = 0; k < TOPK; ++k)
      if (slots[k] < NEXP * CAP) acc += ww[k] * ye[(size_t)slots[k] * H_DIM + h];
    out[(size_t)t * H_DIM + h] = acc;
  }
}

// ---------------- kernel 5: shared experts gate/up -> hs (bf16) ----------

__global__ __launch_bounds__(256)
void shared_gateup(const float* __restrict__ x, const float* __restrict__ sg_w,
                   const float* __restrict__ su_w, bf16_t* __restrict__ hs) {
  __shared__ bf16_t As[2][BM * LDK];
  __shared__ float  Bg[2][BN * LDKF];
  __shared__ float  Bu[2][BN * LDKF];
  const int m0   = blockIdx.y * BM;
  const int n0   = blockIdx.x * BN;
  const int tid  = threadIdx.x;
  const int lane = tid & 31;
  const int w    = tid >> 5;
  const int arow = tid >> 3;
  const int acol = (tid & 7) << 2;

  const float* xrow  = x + (size_t)(m0 + arow) * H_DIM;
  const float* gbase = sg_w + (size_t)n0 * H_DIM;
  const float* ubase = su_w + (size_t)n0 * H_DIM;

  issue_B_async(Bg[0], gbase, H_DIM, 0, w, lane);
  issue_B_async(Bu[0], ubase, H_DIM, 0, w, lane);
  cvt_store4(&As[0][arow * LDK + acol], *(const float4*)(xrow + acol));

  v8f accg0 = {}, accg1 = {}, accu0 = {}, accu1 = {};
  const int KITER = H_DIM / BK;
  for (int it = 0; it < KITER; ++it) {
    const int buf = it & 1;
    const int k0  = it * BK;
    const bool have_next = (it + 1 < KITER);
    float4 areg = make_float4(0.f, 0.f, 0.f, 0.f);
    if (have_next) {
      issue_B_async(Bg[buf ^ 1], gbase, H_DIM, k0 + BK, w, lane);
      issue_B_async(Bu[buf ^ 1], ubase, H_DIM, k0 + BK, w, lane);
      areg = *(const float4*)(xrow + k0 + BK + acol);
    }
    __syncthreads();
    if (have_next) wait_async<8>(); else wait_async<0>();

    v16bf a0 = frag_from_lds(As[buf], 0, lane);
    v16bf a1 = frag_from_lds(As[buf], 16, lane);
    v16bf bg = frag_from_lds_f32(Bg[buf], w * 16, lane);
    v16bf bu = frag_from_lds_f32(Bu[buf], w * 16, lane);
    accg0 = wmma_bf16(a0, bg, accg0);
    accg1 = wmma_bf16(a1, bg, accg1);
    accu0 = wmma_bf16(a0, bu, accu0);
    accu1 = wmma_bf16(a1, bu, accu1);

    if (have_next) cvt_store4(&As[buf ^ 1][arow * LDK + acol], areg);
  }

  const int nCol = n0 + w * 16 + (lane & 15);
  const int rb   = (lane >> 4) * 8;
#pragma unroll
  for (int r = 0; r < 8; ++r) {
    hs[(size_t)(m0 + rb + r)      * FS_DIM + nCol] = (bf16_t)silu_mul(accg0[r], accu0[r]);
    hs[(size_t)(m0 + 16 + rb + r) * FS_DIM + nCol] = (bf16_t)silu_mul(accg1[r], accu1[r]);
  }
}

// ---------------- kernel 6: shared down GEMM, accumulate into out --------

__global__ __launch_bounds__(256)
void shared_down(const bf16_t* __restrict__ hs, const float* __restrict__ sd_w,
                 float* __restrict__ out) {
  __shared__ bf16_t As[2][BM * LDK];
  __shared__ float  Bs[2][BN * LDKF];
  const int m0   = blockIdx.y * BM;
  const int n0   = blockIdx.x * BN;
  const int tid  = threadIdx.x;
  const int lane = tid & 31;
  const int w    = tid >> 5;
  const int arow = tid >> 3;
  const int acol = (tid & 7) << 2;

  const bf16_t* hrow  = hs   + (size_t)(m0 + arow) * FS_DIM;
  const float*  bbase = sd_w + (size_t)n0 * FS_DIM;

  issue_B_async(Bs[0], bbase, FS_DIM, 0, w, lane);
  *(uint2*)&As[0][arow * LDK + acol] = *(const uint2*)(hrow + acol);

  v8f acc0 = {}, acc1 = {};
  const int KITER = FS_DIM / BK;
  for (int it = 0; it < KITER; ++it) {
    const int buf = it & 1;
    const int k0  = it * BK;
    const bool have_next = (it + 1 < KITER);
    uint2 areg = make_uint2(0u, 0u);
    if (have_next) {
      issue_B_async(Bs[buf ^ 1], bbase, FS_DIM, k0 + BK, w, lane);
      areg = *(const uint2*)(hrow + k0 + BK + acol);
    }
    __syncthreads();
    if (have_next) wait_async<4>(); else wait_async<0>();

    v16bf a0 = frag_from_lds(As[buf], 0, lane);
    v16bf a1 = frag_from_lds(As[buf], 16, lane);
    v16bf b  = frag_from_lds_f32(Bs[buf], w * 16, lane);
    acc0 = wmma_bf16(a0, b, acc0);
    acc1 = wmma_bf16(a1, b, acc1);

    if (have_next) *(uint2*)&As[buf ^ 1][arow * LDK + acol] = areg;
  }

  const int n  = n0 + w * 16 + (lane & 15);
  const int rb = (lane >> 4) * 8;
#pragma unroll
  for (int r = 0; r < 8; ++r) {
    out[(size_t)(m0 + rb + r)      * H_DIM + n] += acc0[r];
    out[(size_t)(m0 + 16 + rb + r) * H_DIM + n] += acc1[r];
  }
}

// ---------------- launcher ----------------

extern "C" void kernel_launch(void* const* d_in, const int* in_sizes, int n_in,
                              void* d_out, int out_size, void* d_ws, size_t ws_size,
                              hipStream_t stream) {
  const float* x      = (const float*)d_in[0];
  const float* wg     = (const float*)d_in[1];
  const float* gate_w = (const float*)d_in[2];
  const float* up_w   = (const float*)d_in[3];
  const float* down_w = (const float*)d_in[4];
  const float* sg_w   = (const float*)d_in[5];
  const float* su_w   = (const float*)d_in[6];
  const float* sd_w   = (const float*)d_in[7];
  float* out = (float*)d_out;

  char* ws = (char*)d_ws;
  int*    counts     = (int*)ws;                              // 128 B
  int*    expert_tok = (int*)(ws + 128);                      // 20480 B
  int*    slot_map   = (int*)(ws + 128 + 20480);              // 12288 B
  float*  topkw      = (float*)(ws + 128 + 20480 + 12288);    // 12288 B
  bf16_t* hbuf       = (bf16_t*)(ws + 45184);                 // 14417920 B
  float*  ye         = (float*)(ws + 45184 + 14417920);       // 41943040 B
  bf16_t* hs         = (bf16_t*)(ws + 45184 + 14417920 + 41943040);

  moe_zero_counts<<<1, 32, 0, stream>>>(counts);
  moe_gate<<<T_TOK, 32, 0, stream>>>(x, wg, counts, expert_tok, slot_map, topkw);
  moe_gateup_routed<<<dim3(F_DIM / BN, CAP / BM, NEXP), 256, 0, stream>>>(
      x, gate_w, up_w, counts, expert_tok, hbuf);
  moe_down_routed<<<dim3(H_DIM / BN, CAP / BM, NEXP), 256, 0, stream>>>(
      hbuf, down_w, ye);
  moe_combine<<<T_TOK, 256, 0, stream>>>(ye, slot_map, topkw, out);
  shared_gateup<<<dim3(FS_DIM / BN, T_TOK / BM), 256, 0, stream>>>(x, sg_w, su_w, hs);
  shared_down<<<dim3(H_DIM / BN, T_TOK / BM), 256, 0, stream>>>(hs, sd_w, out);
}
